// vc_interaction_sa_module_86337432584594
// MI455X (gfx1250) — matrix operand order repaired and checked by reading, not compiled, not run on hardware
//
#include <hip/hip_runtime.h>
#include <hip/hip_bf16.h>

// ---------------------------------------------------------------------------
// vc_interaction_sa_module for MI455X (gfx1250, wave32, WMMA)
//
// Pipeline (NB=64 batches, K=512 tokens, C=1024, ATT=128, MID=512):
//   theta/phi/psi = x @ W^T + b          (bf16 WMMA, fp32 acc)
//   cov   = phi @ theta^T                (per-batch, fp32 out)
//   cov   = softmax(cov, axis=1)         (online column softmax, fp32 -> bf16)
//   o     = cov @ psi                    (bf16 WMMA)
//   h     = relu(o @ W1^T + b1)          (bf16 WMMA)
//   out   = x + gamma * (h @ W2^T + b2)  (bf16 WMMA, fp32 epilogue)
//
// GEMM: one wave owns 1 M-tile x NT(=4) N-tiles of 16x16, and the K loop is
// register double-buffered: fragments of chunk k+1 are loaded into a second
// register set while the 4-WMMA burst consumes chunk k, forcing distinct
// registers and partial s_wait_loadcnt instead of full drains.
// Requires K % 64 == 0 (true for all GEMMs here: 1024/512/128).
// ---------------------------------------------------------------------------

typedef __attribute__((ext_vector_type(16))) __bf16 v16bf;
typedef __attribute__((ext_vector_type(8)))  __bf16 v8bf;
typedef __attribute__((ext_vector_type(4)))  __bf16 v4bf;
typedef __attribute__((ext_vector_type(8)))  float  v8f;

union V16U { v16bf v16; v8bf v8[2]; };

__device__ __forceinline__ __bf16 to_bf16(float f) {
    // round-to-nearest-even, bit-level
    unsigned u = __float_as_uint(f);
    unsigned r = (u + 0x7FFFu + ((u >> 16) & 1u)) >> 16;
    unsigned short h = (unsigned short)r;
    __bf16 b;
    __builtin_memcpy(&b, &h, 2);
    return b;
}

// ------------------------- epilogue modes ----------------------------------
constexpr int EPI_BF16           = 0; // bf16 row-major, no bias
constexpr int EPI_BF16_BIAS      = 1; // bf16 row-major, + bias
constexpr int EPI_BF16_BIAS_T    = 2; // bf16 TRANSPOSED store, + bias
constexpr int EPI_F32            = 3; // fp32 row-major, no bias
constexpr int EPI_BF16_BIAS_RELU = 4; // bf16 row-major, + bias, relu
constexpr int EPI_RESID          = 5; // fp32: X + gamma*(acc + bias)

// ---------------------------------------------------------------------------
// Batched bf16 GEMM:  D[n] = A[n] (MxK, row-major, lda) x Bt[n]^T
// (Bt is "N-major": row per output column, contiguous K, ldb).
// block = (32 lanes, 8 waves); grid = (Mtiles/8, Ntiles/NT, NB)
// ---------------------------------------------------------------------------
template <int MODE, int NT>
__global__ __launch_bounds__(256) void wmma_gemm_bf16(
    const __bf16* __restrict__ A,  long sAn, int lda,
    const __bf16* __restrict__ Bt, long sBn, int ldb,
    void* __restrict__ Dv,         long sDn, int ldd,
    int M, int K,
    const float* __restrict__ bias,
    const float* __restrict__ X,
    const float* __restrict__ gamma)
{
    const int lane  = threadIdx.x;                         // 0..31
    const int wave  = threadIdx.y;                         // 0..7
    const int bn    = blockIdx.z;
    const int tileM = (blockIdx.x * blockDim.y + wave) * 16;
    const int tileN = blockIdx.y * (16 * NT);
    if (tileM >= M) return;                                // wave-uniform

    // CDNA5 WMMA wave32 fragment addressing:
    //  A 16x32 bf16: lane L holds row (L&15); K = akb+[0..7], akb+16+[0..7],
    //                akb = 0 (lanes 0-15) or 8 (lanes 16-31).
    //  B 32x16 bf16: lane L holds col (L&15); K = bkb+[0..7], bkb+16+[0..7],
    //                bkb = 0 (lanes 0-15) or 8 (lanes 16-31).
    const __bf16* arow  = A  + (long)bn * sAn + (long)(tileM + (lane & 15)) * lda;
    const __bf16* bcol0 = Bt + (long)bn * sBn + (long)(tileN + (lane & 15)) * ldb;
    const int akb = (lane & 16) ? 8  : 0;
    const int bkb = (lane & 16) ? 8  : 0;

    v8f acc[NT];
    #pragma unroll
    for (int nt = 0; nt < NT; ++nt) acc[nt] = (v8f){};

    auto load_chunk = [&](int kk, V16U& av, V16U (&bv)[NT]) {
        av.v8[0] = *(const v8bf*)(arow + kk + akb);
        av.v8[1] = *(const v8bf*)(arow + kk + akb + 16);
        #pragma unroll
        for (int nt = 0; nt < NT; ++nt) {
            const __bf16* bc = bcol0 + (long)nt * 16 * ldb;
            bv[nt].v8[0] = *(const v8bf*)(bc + kk + bkb);
            bv[nt].v8[1] = *(const v8bf*)(bc + kk + bkb + 16);
        }
    };
    auto mma_burst = [&](const V16U& av, const V16U (&bv)[NT]) {
        #pragma unroll
        for (int nt = 0; nt < NT; ++nt) {
            acc[nt] = __builtin_amdgcn_wmma_f32_16x16x32_bf16(
                /*neg_a=*/false, av.v16, /*neg_b=*/false, bv[nt].v16,
                /*c_mod=*/(short)0, acc[nt], /*reuse_a=*/false, /*reuse_b=*/false);
        }
    };

    // 2-deep register pipeline over 32-wide K chunks (K % 64 == 0).
    V16U avA, bvA[NT], avB, bvB[NT];
    load_chunk(0, avA, bvA);
    for (int kk = 0; kk < K; kk += 64) {
        __builtin_prefetch(arow + kk + 256, 0, 3);   // A stream, 4 iterations (256 k) ahead
        load_chunk(kk + 32, avB, bvB);               // next chunk in flight
        mma_burst(avA, bvA);                         // consume current chunk
        if (kk + 64 < K) load_chunk(kk + 64, avA, bvA);
        mma_burst(avB, bvB);
    }

    // C/D layout: VGPR r -> row rbase+r, column = lane&15
    const int col   = lane & 15;
    const int rbase = (lane & 16) ? 8 : 0;

    #pragma unroll
    for (int nt = 0; nt < NT; ++nt) {
        const int ncol = tileN + nt * 16 + col;
        float bvv = 0.0f;
        if constexpr (MODE == EPI_BF16_BIAS || MODE == EPI_BF16_BIAS_T ||
                      MODE == EPI_BF16_BIAS_RELU || MODE == EPI_RESID)
            bvv = bias[ncol];

        if constexpr (MODE == EPI_BF16 || MODE == EPI_BF16_BIAS ||
                      MODE == EPI_BF16_BIAS_RELU) {
            __bf16* d = (__bf16*)Dv + (long)bn * sDn;
            #pragma unroll
            for (int r = 0; r < 8; ++r) {
                float v = acc[nt][r] + bvv;
                if constexpr (MODE == EPI_BF16_BIAS_RELU) v = fmaxf(v, 0.0f);
                d[(long)(tileM + rbase + r) * ldd + ncol] = to_bf16(v);
            }
        } else if constexpr (MODE == EPI_BF16_BIAS_T) {
            // D[col][row]; 8 consecutive rows per lane -> one 16B store
            __bf16* d = (__bf16*)Dv + (long)bn * sDn;
            v8bf o;
            #pragma unroll
            for (int r = 0; r < 8; ++r) o[r] = to_bf16(acc[nt][r] + bvv);
            *(v8bf*)(d + (long)ncol * ldd + tileM + rbase) = o;
        } else if constexpr (MODE == EPI_F32) {
            float* d = (float*)Dv + (long)bn * sDn;
            #pragma unroll
            for (int r = 0; r < 8; ++r)
                d[(long)(tileM + rbase + r) * ldd + ncol] = acc[nt][r];
        } else { // EPI_RESID
            float* d = (float*)Dv + (long)bn * sDn;
            const float g = gamma[0];
            #pragma unroll
            for (int r = 0; r < 8; ++r) {
                long idx = (long)(tileM + rbase + r) * ldd + ncol;
                d[idx] = X[(long)bn * sDn + idx] + g * (acc[nt][r] + bvv);
            }
        }
    }
}

// --------------------------- fp32 -> bf16 (x4 vectorized) -------------------
__global__ __launch_bounds__(256) void f32_to_bf16_kernel(
    const float* __restrict__ in, __bf16* __restrict__ out, long n4)
{
    long i = (long)blockIdx.x * blockDim.x + threadIdx.x;
    if (i < n4) {
        float4 f = ((const float4*)in)[i];
        v4bf o;
        o[0] = to_bf16(f.x); o[1] = to_bf16(f.y);
        o[2] = to_bf16(f.z); o[3] = to_bf16(f.w);
        ((v4bf*)out)[i] = o;
    }
}

// ------------- online column softmax over axis=1: cov[n, :, j] --------------
// cov fp32 [NB,512,512] -> covS bf16 [NB,512,512]. One thread per column j;
// consecutive threads read consecutive j => coalesced row reads. Two passes.
__global__ __launch_bounds__(256) void col_softmax_kernel(
    const float* __restrict__ cov, __bf16* __restrict__ covS)
{
    const int j = blockIdx.x * blockDim.x + threadIdx.x;   // 0..511
    const int n = blockIdx.y;
    const float* c = cov  + (long)n * 512 * 512;
    __bf16*      o = covS + (long)n * 512 * 512;

    float m = -3.402823466e+38f, s = 0.0f;
    for (int i = 0; i < 512; ++i) {
        float v  = c[(long)i * 512 + j];
        float mn = fmaxf(m, v);
        s = s * __expf(m - mn) + __expf(v - mn);
        m = mn;
    }
    const float inv = 1.0f / s;
    for (int i = 0; i < 512; ++i)
        o[(long)i * 512 + j] = to_bf16(__expf(c[(long)i * 512 + j] - m) * inv);
}

// ---------------------------------------------------------------------------
extern "C" void kernel_launch(void* const* d_in, const int* in_sizes, int n_in,
                              void* d_out, int out_size, void* d_ws, size_t ws_size,
                              hipStream_t stream)
{
    (void)in_sizes; (void)n_in; (void)out_size; (void)ws_size;

    const float* x       = (const float*)d_in[0];   // [64,512,1024]
    const float* W_theta = (const float*)d_in[1];   // [128,1024]
    const float* b_theta = (const float*)d_in[2];
    const float* W_phi   = (const float*)d_in[3];
    const float* b_phi   = (const float*)d_in[4];
    const float* W_psi   = (const float*)d_in[5];
    const float* b_psi   = (const float*)d_in[6];
    const float* W1      = (const float*)d_in[7];   // [512,128]
    const float* b1      = (const float*)d_in[8];
    const float* W2      = (const float*)d_in[9];   // [1024,512]
    const float* b2      = (const float*)d_in[10];
    const float* gamma   = (const float*)d_in[11];
    float* out           = (float*)d_out;           // [64,512,1024]

    const long NB = 64, KT = 512, C = 1024, ATT = 128, MID = 512;
    const long R = NB * KT;                         // 32768 global rows
    constexpr int NT = 4;                           // N-tiles per wave

    // ---------------- workspace carve-up (~237 MB) ----------------
    char* ws = (char*)d_ws;
    size_t off = 0;
    auto take = [&](size_t bytes) -> void* {
        void* p = ws + off;
        off += (bytes + 255) & ~(size_t)255;
        return p;
    };
    __bf16* xb    = (__bf16*)take(R * C * 2);         // x bf16
    __bf16* wthb  = (__bf16*)take(ATT * C * 2);
    __bf16* wphb  = (__bf16*)take(ATT * C * 2);
    __bf16* wpsb  = (__bf16*)take(ATT * C * 2);
    __bf16* w1b   = (__bf16*)take(MID * ATT * 2);
    __bf16* w2b   = (__bf16*)take(C * MID * 2);
    __bf16* thb   = (__bf16*)take(R * ATT * 2);       // theta  [R,128]
    __bf16* phb   = (__bf16*)take(R * ATT * 2);       // phi    [R,128]
    __bf16* psTb  = (__bf16*)take(ATT * R * 2);       // psi^T  [128,R]
    float*  cov   = (float*) take(NB * KT * KT * 4);  // [64,512,512] fp32
    __bf16* covS  = (__bf16*)take(NB * KT * KT * 2);  // softmax bf16
    __bf16* ob    = (__bf16*)take(R * ATT * 2);       // o      [R,128]
    __bf16* hb    = (__bf16*)take(R * MID * 2);       // h      [R,512]

    const dim3 blk(32, 8);   // 8 waves; each owns 1 M-tile x 4 N-tiles

    // ---------------- fp32 -> bf16 conversions ----------------
    auto cvt = [&](const float* src, __bf16* dst, long n) {
        long n4 = n / 4;
        f32_to_bf16_kernel<<<dim3((unsigned)((n4 + 255) / 256)), 256, 0, stream>>>(src, dst, n4);
    };
    cvt(x,       xb,   R * C);
    cvt(W_theta, wthb, ATT * C);
    cvt(W_phi,   wphb, ATT * C);
    cvt(W_psi,   wpsb, ATT * C);
    cvt(W1,      w1b,  MID * ATT);
    cvt(W2,      w2b,  C * MID);

    // ---------------- projections: [32768x1024] x [128x1024]^T ----------------
    {
        dim3 grid((unsigned)(R / 16 / 8), (unsigned)(ATT / 16 / NT), 1);
        wmma_gemm_bf16<EPI_BF16_BIAS, NT><<<grid, blk, 0, stream>>>(
            xb, 0, (int)C, wthb, 0, (int)C, thb, 0, (int)ATT,
            (int)R, (int)C, b_theta, nullptr, nullptr);
        wmma_gemm_bf16<EPI_BF16_BIAS, NT><<<grid, blk, 0, stream>>>(
            xb, 0, (int)C, wphb, 0, (int)C, phb, 0, (int)ATT,
            (int)R, (int)C, b_phi, nullptr, nullptr);
        // psi stored transposed [128][32768] so it is B-ready for the o GEMM
        wmma_gemm_bf16<EPI_BF16_BIAS_T, NT><<<grid, blk, 0, stream>>>(
            xb, 0, (int)C, wpsb, 0, (int)C, psTb, 0, (int)R,
            (int)R, (int)C, b_psi, nullptr, nullptr);
    }

    // ---------------- cov[n] = phi[n] @ theta[n]^T : [512x512], K=128 ----------
    {
        dim3 grid((unsigned)(KT / 16 / 8), (unsigned)(KT / 16 / NT), (unsigned)NB);
        wmma_gemm_bf16<EPI_F32, NT><<<grid, blk, 0, stream>>>(
            phb, KT * ATT, (int)ATT, thb, KT * ATT, (int)ATT,
            cov, KT * KT, (int)KT,
            (int)KT, (int)ATT, nullptr, nullptr, nullptr);
    }

    // ---------------- softmax over axis=1 (columns), fp32 -> bf16 --------------
    {
        dim3 grid((unsigned)(KT / 256), (unsigned)NB);
        col_softmax_kernel<<<grid, 256, 0, stream>>>(cov, covS);
    }

    // ---------------- o[n] = covS[n] @ psi[n] : [512x128], K=512 ---------------
    {
        // B for batch n is psi^T columns n*512..n*512+511 -> base offset n*512
        dim3 grid((unsigned)(KT / 16 / 8), (unsigned)(ATT / 16 / NT), (unsigned)NB);
        wmma_gemm_bf16<EPI_BF16, NT><<<grid, blk, 0, stream>>>(
            covS, KT * KT, (int)KT, psTb, KT /*elem offset per n*/, (int)R,
            ob, KT * ATT, (int)ATT,
            (int)KT, (int)KT, nullptr, nullptr, nullptr);
    }

    // ---------------- h = relu(o @ W1^T + b1) : [32768x512], K=128 -------------
    {
        dim3 grid((unsigned)(R / 16 / 8), (unsigned)(MID / 16 / NT), 1);
        wmma_gemm_bf16<EPI_BF16_BIAS_RELU, NT><<<grid, blk, 0, stream>>>(
            ob, 0, (int)ATT, w1b, 0, (int)ATT, hb, 0, (int)MID,
            (int)R, (int)ATT, b1, nullptr, nullptr);
    }

    // ---------------- out = x + gamma*(h @ W2^T + b2) : [32768x1024] -----------
    {
        dim3 grid((unsigned)(R / 16 / 8), (unsigned)(C / 16 / NT), 1);
        wmma_gemm_bf16<EPI_RESID, NT><<<grid, blk, 0, stream>>>(
            hb, 0, (int)MID, w2b, 0, (int)MID, out, 0, (int)C,
            (int)R, (int)MID, b2, x, gamma);
    }
}
